// SupervisedContrastiveLoss_6279242187472
// MI455X (gfx1250) — matrix operand order repaired and checked by reading, not compiled, not run on hardware
//
#include <hip/hip_runtime.h>
#include <hip/hip_bf16.h>

typedef __attribute__((ext_vector_type(16))) _Float16 v16h;
typedef __attribute__((ext_vector_type(8)))  _Float16 v8h;
typedef __attribute__((ext_vector_type(2)))  _Float16 v2h;
typedef __attribute__((ext_vector_type(8)))  float    v8f;

#define TEMP_INV  10.0f      // 1 / TEMPERATURE
#define LOGIT_MAX 10.0f      // |dot| <= 1 (unit vectors) -> |logit| <= 10: fixed softmax max
#define EPSF      1e-8f

constexpr int B_  = 2;
constexpr int V_  = 2;
constexpr int C_  = 64;
constexpr int HW_ = 1024;
constexpr int N_  = B_ * V_ * HW_;   // 4096 anchors
constexpr int IMG = V_ * HW_;        // 2048 rows per image (only same-image pairs matter)

// ---------------------------------------------------------------------------
// Kernel 0: zero the global accumulators (must run every call: graph replay)
// ---------------------------------------------------------------------------
__global__ void scl_zero(float* __restrict__ acc) {
    if (threadIdx.x < 4) acc[threadIdx.x] = 0.0f;
}

// ---------------------------------------------------------------------------
// Kernel 1: [B,V,C,HW] -> [N,64], L2-normalize, convert to f16.
// One wave32 per row; lane handles 2 channels (stride HW gather).
// ---------------------------------------------------------------------------
__global__ void scl_normalize(const float* __restrict__ stu,
                              const float* __restrict__ tea,
                              _Float16* __restrict__ s16,
                              _Float16* __restrict__ t16) {
    const int wid  = threadIdx.x >> 5;
    const int lane = threadIdx.x & 31;
    const int r    = blockIdx.x * 8 + wid;        // 0 .. 2N-1

    const float* src; _Float16* dst; int row;
    if (r < N_) { src = stu; dst = s16; row = r; }
    else        { src = tea; dst = t16; row = r - N_; }

    const int bv = row >> 10;        // row / HW
    const int hw = row & (HW_ - 1);
    const float* base = src + (size_t)bv * C_ * HW_ + hw;

    const int c  = lane * 2;
    const float v0 = base[(size_t)c       * HW_];
    const float v1 = base[(size_t)(c + 1) * HW_];

    float ss = v0 * v0 + v1 * v1;
    #pragma unroll
    for (int off = 16; off > 0; off >>= 1) ss += __shfl_xor(ss, off);

    const float inv = 1.0f / fmaxf(__builtin_sqrtf(ss), 1e-12f);
    v2h p;
    p[0] = (_Float16)(v0 * inv);
    p[1] = (_Float16)(v1 * inv);
    *(v2h*)(dst + (size_t)row * C_ + c) = p;
}

// ---------------------------------------------------------------------------
// Kernel 2: WMMA logits + branchless register-resident softmax accumulation.
// grid = N/16 blocks; block = 4 waves; wave w covers col-tiles [w*32, w*32+32)
// of this anchor tile's own image. Fixed softmax max (=10) => no rescale, no
// LDS tile spill; all masking is data-flow (cndmask), not EXEC control flow.
// ---------------------------------------------------------------------------
__global__ void scl_main(const _Float16* __restrict__ s16,
                         const _Float16* __restrict__ t16,
                         const int*      __restrict__ labels,
                         float*          __restrict__ acc) {
    __shared__ float sM[4][16], nsM[4][16], ncM[4][16];

    const int w    = threadIdx.x >> 5;
    const int lane = threadIdx.x & 31;
    const int half = lane >> 4;       // 0: lanes 0-15, 1: lanes 16-31
    const int l16  = lane & 15;

    const int rowBase = blockIdx.x * 16;
    const int img     = rowBase / IMG;
    const int colImg  = img * IMG;
    const int myRow0  = rowBase + half * 8;   // global row of C-fragment reg r=0

    // ---- A fragments (16x32 f16 layout, ISA 7.12.2), K = 0..31 and 32..63 ----
    const _Float16* ap = s16 + (size_t)(rowBase + l16) * C_;
    v16h a0, a1;
    {
        v8h lo0 = *(const v8h*)(ap +      half * 8);
        v8h hi0 = *(const v8h*)(ap + 16 + half * 8);
        a0 = __builtin_shufflevector(lo0, hi0, 0,1,2,3,4,5,6,7,8,9,10,11,12,13,14,15);
        v8h lo1 = *(const v8h*)(ap + 32 + half * 8);
        v8h hi1 = *(const v8h*)(ap + 48 + half * 8);
        a1 = __builtin_shufflevector(lo1, hi1, 0,1,2,3,4,5,6,7,8,9,10,11,12,13,14,15);
    }

    // labels of my 8 rows (C-fragment rows half*8 + r); 32B aligned int4 pair
    int rowLabR[8];
    {
        const int4* lp = (const int4*)(labels + myRow0);
        const int4 la = lp[0], lb = lp[1];
        rowLabR[0] = la.x; rowLabR[1] = la.y; rowLabR[2] = la.z; rowLabR[3] = la.w;
        rowLabR[4] = lb.x; rowLabR[5] = lb.y; rowLabR[6] = lb.z; rowLabR[7] = lb.w;
    }

    // per-lane partial stats: row = half*8 + r, this lane's column only
    float sAcc[8], nsAcc[8], ncAcc[8];
    #pragma unroll
    for (int r = 0; r < 8; ++r) { sAcc[r] = 0.0f; nsAcc[r] = 0.0f; ncAcc[r] = 0.0f; }

    // B-fragment loader (32x16 f16 layout: lane = col, K in table order, same as A)
    auto loadB = [&](int tt, v16h& b0, v16h& b1, int& cl) {
        const int colBase = colImg + (w * 32 + tt) * 16;
        const _Float16* bp = t16 + (size_t)(colBase + l16) * C_;
        v8h b0lo = *(const v8h*)(bp +      half * 8);
        v8h b0hi = *(const v8h*)(bp + 16 + half * 8);
        b0 = __builtin_shufflevector(b0lo, b0hi, 0,1,2,3,4,5,6,7,8,9,10,11,12,13,14,15);
        v8h b1lo = *(const v8h*)(bp + 32 + half * 8);
        v8h b1hi = *(const v8h*)(bp + 48 + half * 8);
        b1 = __builtin_shufflevector(b1lo, b1hi, 0,1,2,3,4,5,6,7,8,9,10,11,12,13,14,15);
        cl = labels[colBase + l16];
    };

    v16h b0c, b1c; int clc;
    loadB(0, b0c, b1c, clc);

    for (int tt = 0; tt < 32; ++tt) {
        // prefetch next tile while this tile computes
        v16h b0n, b1n; int cln = 0;
        if (tt + 1 < 32) loadB(tt + 1, b0n, b1n, cln);

        v8f cacc = {};
        cacc = __builtin_amdgcn_wmma_f32_16x16x32_f16(false, a0, false, b0c,
                                                      (short)0, cacc, false, false);
        cacc = __builtin_amdgcn_wmma_f32_16x16x32_f16(false, a1, false, b1c,
                                                      (short)0, cacc, false, false);

        const int myCol = colImg + (w * 32 + tt) * 16 + l16;  // global column j
        const int diag  = myCol - myRow0;                     // self pair when diag == r

        #pragma unroll
        for (int r = 0; r < 8; ++r) {
            const float x       = cacc[r] * TEMP_INV;                 // logit
            const float notself = (diag == r) ? 0.0f : 1.0f;          // v_cndmask
            const float e       = notself * __expf(x - LOGIT_MAX);
            sAcc[r] += e;                                             // denominator
            const float match   = (clc == rowLabR[r]) ? notself : 0.0f;
            nsAcc[r] = __builtin_fmaf(match, x, nsAcc[r]);            // numerator sum
            ncAcc[r] += match;                                        // numerator count
        }

        b0c = b0n; b1c = b1n; clc = cln;
    }

    // collapse the 16 columns of each half-wave (masks < 16 stay in-group)
    #pragma unroll
    for (int r = 0; r < 8; ++r) {
        #pragma unroll
        for (int off = 8; off > 0; off >>= 1) {
            sAcc[r]  += __shfl_xor(sAcc[r],  off);
            nsAcc[r] += __shfl_xor(nsAcc[r], off);
            ncAcc[r] += __shfl_xor(ncAcc[r], off);
        }
    }

    if (l16 == 0) {   // lane 0 -> rows 0..7, lane 16 -> rows 8..15
        #pragma unroll
        for (int r = 0; r < 8; ++r) {
            sM [w][half * 8 + r] = sAcc[r];
            nsM[w][half * 8 + r] = nsAcc[r];
            ncM[w][half * 8 + r] = ncAcc[r];
        }
    }
    __syncthreads();

    // merge the 4 waves' partial stats and emit per-anchor results
    if (w == 0 && lane < 16) {
        float S = 0.0f, NS = 0.0f, NC = 0.0f;
        #pragma unroll
        for (int q = 0; q < 4; ++q) { S += sM[q][lane]; NS += nsM[q][lane]; NC += ncM[q][lane]; }
        if (NC > 0.5f) {  // valid anchor: at least one positive
            // sum_num(log_prob) = NS - NC*(LOGIT_MAX + log S); mean over (NC+eps)
            const float mlp = (NS - NC * (LOGIT_MAX + __logf(S))) / (NC + EPSF);
            atomicAdd(&acc[0], mlp);
            atomicAdd(&acc[1], 1.0f);
            if (labels[rowBase + lane] != 0) atomicAdd(&acc[2], 1.0f);
        }
    }
}

// ---------------------------------------------------------------------------
// Kernel 3: finalize scalar loss.
// ---------------------------------------------------------------------------
__global__ void scl_final(const float* __restrict__ acc, float* __restrict__ out) {
    const float sum = acc[0], nv = acc[1], nbg = acc[2];
    const float loss = (nv > 0.0f) ? (-sum / nv) : 0.0f;
    out[0] = loss * nbg / (nbg + EPSF);
}

// ---------------------------------------------------------------------------
extern "C" void kernel_launch(void* const* d_in, const int* in_sizes, int n_in,
                              void* d_out, int out_size, void* d_ws, size_t ws_size,
                              hipStream_t stream) {
    const float* stu    = (const float*)d_in[0];
    const float* tea    = (const float*)d_in[1];
    const int*   labels = (const int*)d_in[2];
    float*       out    = (float*)d_out;

    char* ws = (char*)d_ws;
    float*    acc = (float*)ws;                                   // 4 floats
    _Float16* s16 = (_Float16*)(ws + 256);                        // 4096*64 f16
    _Float16* t16 = (_Float16*)(ws + 256 + (size_t)N_ * C_ * 2);  // 4096*64 f16

    scl_zero     <<<1,            32,  0, stream>>>(acc);
    scl_normalize<<<(2 * N_) / 8, 256, 0, stream>>>(stu, tea, s16, t16);
    scl_main     <<<N_ / 16,      128, 0, stream>>>(s16, t16, labels, acc);
    scl_final    <<<1,            1,   0, stream>>>(acc, out);
}
